// TreeLSTM_61418032333368
// MI455X (gfx1250) — matrix operand order repaired and checked by reading, not compiled, not run on hardware
//
#include <hip/hip_runtime.h>

typedef _Float16 v16h __attribute__((ext_vector_type(16)));
typedef _Float16 v8h  __attribute__((ext_vector_type(8)));
typedef float    v8f  __attribute__((ext_vector_type(8)));

#define HDIM    128
#define NLEAF   65536
#define NCLS    5
#define TOTROWS 131071   // 65536 + 32768 + ... + 1

// staged kernels: 8 waves (= 8 M-tiles) per block, sharing LDS-staged weights
#define SWAVES  8
#define SBLOCK  (SWAVES * 32)
// projection kernel: light-weight, 4 waves per block
#define PWPB    4
#define PBLOCK  (PWPB * 32)

// levels 1..TAIL_FIRST-1 run as separate launches; levels TAIL_FIRST..16 are
// fused into one single-workgroup launch (ntiles <= SWAVES there).
#define TAIL_FIRST 9

__device__ __forceinline__ float sigf(float x) { return 1.0f / (1.0f + __expf(-x)); }

#define WMMA_F16(A, B, C) \
  __builtin_amdgcn_wmma_f32_16x16x32_f16(false, (A), false, (B), (short)0, (C), false, false)

// ---------------- fragment helpers ----------------

// Direct-from-global B fragment (projection kernel):
// lane l holds B[kb..kb+15][n], n = l&15, kb = kc*32 + (l>>4)*16.
// Weights pre-transposed to wT[n*rowstride + k] (N-major, K contiguous).
__device__ __forceinline__ v16h load_b_frag(const _Float16* __restrict__ w,
                                            int ncol, int kbase, int rowstride) {
  return *(const v16h*)(w + (size_t)ncol * rowstride + kbase);
}

// A fragment from f16 row: lane holds row[k0..k0+7] and row[k0+16..k0+23],
// k0 = kc*32 + (lane>>4)*8  (ISA 16-bit A 16x32 layout).
__device__ __forceinline__ v16h load_a_f16(const _Float16* __restrict__ hrow, int k0) {
  v8h lo = *(const v8h*)(hrow + k0);
  v8h hi = *(const v8h*)(hrow + k0 + 16);
  v16h r;
#pragma unroll
  for (int j = 0; j < 8; ++j) { r[j] = lo[j]; r[8 + j] = hi[j]; }
  return r;
}

__device__ __forceinline__ v16h load_a_f32(const float* __restrict__ erow, int k0) {
  v8f lo = *(const v8f*)(erow + k0);
  v8f hi = *(const v8f*)(erow + k0 + 16);
  v16h r;
#pragma unroll
  for (int j = 0; j < 8; ++j) { r[j] = (_Float16)lo[j]; r[8 + j] = (_Float16)hi[j]; }
  return r;
}

// LDS-staged B fragment. LDS layout: fragment f = 64 chunks of 16B;
// chunk id within frag = h2*32 + lane -> lane reads at 16B stride
// (4-dword lane stride covers all 64 banks in the minimum passes).
__device__ __forceinline__ v16h lds_b_frag(const _Float16* smem, int f, int lane) {
  v8h lo = *(const v8h*)(smem + (size_t)(f * 64 + lane) * 8);
  v8h hi = *(const v8h*)(smem + (size_t)(f * 64 + 32 + lane) * 8);
  v16h r;
#pragma unroll
  for (int j = 0; j < 8; ++j) { r[j] = lo[j]; r[8 + j] = hi[j]; }
  return r;
}

// ---- weight transpose/convert: dst[n*K + k] = (f16) src[k*N + n], zero-pad n>=N ----
__global__ void transpose_pad_kernel(const float* __restrict__ src, _Float16* __restrict__ dst,
                                     int K, int N, int Npad) {
  int idx = blockIdx.x * blockDim.x + threadIdx.x;
  if (idx >= K * Npad) return;
  int n = idx / K, k = idx - n * K;
  dst[idx] = (n < N) ? (_Float16)src[(size_t)k * N + n] : (_Float16)0.0f;
}

// ---- leaf stage: e = emb[words]; i,o,u = act(e@W+b); c = i*u; h = o*relu(c) ----
__global__ void __launch_bounds__(SBLOCK)
tree_leaf_kernel(const int* __restrict__ words, const float* __restrict__ emb,
                 const _Float16* __restrict__ WiT, const _Float16* __restrict__ WoT,
                 const _Float16* __restrict__ WuT,
                 const float* __restrict__ bi, const float* __restrict__ bo,
                 const float* __restrict__ bu,
                 _Float16* __restrict__ h_base, float* __restrict__ c_base) {
  __shared__ _Float16 smem[12 * 512];   // 12 frags * 1KB per ntile stage
  int lane  = threadIdx.x & 31;
  int wave  = threadIdx.x >> 5;
  int tile  = blockIdx.x * SWAVES + wave;
  bool active = (tile < NLEAF / 16);
  int row0  = (active ? tile : 0) << 4;
  int m     = lane & 15;
  int khalf = lane >> 4;

  int wid = words[row0 + m];
  const float* erow = emb + (size_t)wid * HDIM;
  v16h a[4];
#pragma unroll
  for (int kc = 0; kc < 4; ++kc) a[kc] = load_a_f32(erow, kc * 32 + khalf * 8);

  for (int nt = 0; nt < 8; ++nt) {
    __syncthreads();
    for (int cid = threadIdx.x; cid < 12 * 64; cid += SBLOCK) {
      int f  = cid >> 6;
      int r  = cid & 63;
      int h2 = r >> 5;
      int l  = r & 31;
      int kc = f / 3, wsel = f % 3;
      int n  = (nt << 4) + (l & 15);
      int kb = kc * 32 + ((l >> 4) << 3) + h2 * 16;
      const _Float16* src = (wsel == 0 ? WiT : (wsel == 1 ? WoT : WuT)) + n * HDIM + kb;
      *(v8h*)(smem + (size_t)cid * 8) = *(const v8h*)src;
    }
    __syncthreads();

    if (active) {
      int ncol = (nt << 4) + m;
      v8f ai = {}, ao = {}, au = {};
#pragma unroll
      for (int kc = 0; kc < 4; ++kc) {
        ai = WMMA_F16(a[kc], lds_b_frag(smem, kc * 3 + 0, lane), ai);
        ao = WMMA_F16(a[kc], lds_b_frag(smem, kc * 3 + 1, lane), ao);
        au = WMMA_F16(a[kc], lds_b_frag(smem, kc * 3 + 2, lane), au);
      }
      float vbi = bi[ncol], vbo = bo[ncol], vbu = bu[ncol];
#pragma unroll
      for (int r = 0; r < 8; ++r) {
        int row = row0 + r + khalf * 8;
        float iv = sigf(ai[r] + vbi);
        float ov = sigf(ao[r] + vbo);
        float uv = fmaxf(au[r] + vbu, 0.0f);
        float cv = iv * uv;
        float hv = ov * fmaxf(cv, 0.0f);
        c_base[(size_t)row * HDIM + ncol] = cv;
        h_base[(size_t)row * HDIM + ncol] = (_Float16)hv;
      }
    }
  }
}

// ---- shared per-level tile body: fused 5 GEMMs + cell update, LDS-staged B ----
// Must be executed by ALL threads of the block (contains barriers).
__device__ __forceinline__ void
level_tile_body(_Float16* smem, _Float16* __restrict__ h_base, float* __restrict__ c_base,
                int off_prev, int off_cur, int nrows, int tile,
                const _Float16* __restrict__ UiT, const _Float16* __restrict__ UoT,
                const _Float16* __restrict__ UuT,
                const _Float16* __restrict__ Uf1T, const _Float16* __restrict__ Uf2T,
                const float* __restrict__ bUi, const float* __restrict__ bUo,
                const float* __restrict__ bUu,
                const float* __restrict__ bf1, const float* __restrict__ bf2) {
  int lane   = threadIdx.x & 31;
  int ntiles = (nrows + 15) >> 4;
  bool active = (tile < ntiles);
  int row0  = (active ? tile : 0) << 4;
  int m     = lane & 15;
  int khalf = lane >> 4;

  int arow = min(row0 + m, nrows - 1);   // clamp so EXEC stays full
  const _Float16* hl = h_base + (size_t)(off_prev + 2 * arow) * HDIM;
  const _Float16* hr = hl + HDIM;
  v16h aL[4], aR[4];
#pragma unroll
  for (int kc = 0; kc < 4; ++kc) {
    int k0 = kc * 32 + khalf * 8;
    aL[kc] = load_a_f16(hl, k0);
    aR[kc] = load_a_f16(hr, k0);
  }
  const float* c_prev = c_base + (size_t)off_prev * HDIM;
  float*       c_out  = c_base + (size_t)off_cur  * HDIM;
  _Float16*    h_out  = h_base + (size_t)off_cur  * HDIM;

  for (int nt = 0; nt < 8; ++nt) {
    __syncthreads();
    // stage 32 frags: (Ui_t, Ui_b, Uo_t, Uo_b, Uu_t, Uu_b, Uf1, Uf2) x 4 k-chunks
#pragma unroll
    for (int j = 0; j < (32 * 64) / SBLOCK; ++j) {
      int cid = j * SBLOCK + threadIdx.x;
      int f  = cid >> 6;
      int r  = cid & 63;
      int h2 = r >> 5;
      int l  = r & 31;
      int kc = f >> 3, wsel = f & 7;
      int n  = (nt << 4) + (l & 15);
      int kb = kc * 32 + ((l >> 4) << 3) + h2 * 16;
      const _Float16* src;
      switch (wsel) {
        case 0:  src = UiT  + n * 256 + kb;       break;
        case 1:  src = UiT  + n * 256 + kb + 128; break;
        case 2:  src = UoT  + n * 256 + kb;       break;
        case 3:  src = UoT  + n * 256 + kb + 128; break;
        case 4:  src = UuT  + n * 256 + kb;       break;
        case 5:  src = UuT  + n * 256 + kb + 128; break;
        case 6:  src = Uf1T + n * 128 + kb;       break;
        default: src = Uf2T + n * 128 + kb;       break;
      }
      *(v8h*)(smem + (size_t)cid * 8) = *(const v8h*)src;
    }
    __syncthreads();

    if (active) {
      int ncol = (nt << 4) + m;
      v8f ai = {}, ao = {}, au = {}, a1 = {}, a2 = {};
#pragma unroll
      for (int kc = 0; kc < 4; ++kc) {
        int fb = kc * 8;
        // eN@Ui = lH@Ui[0:128] + rH@Ui[128:256]
        ai = WMMA_F16(aL[kc], lds_b_frag(smem, fb + 0, lane), ai);
        ai = WMMA_F16(aR[kc], lds_b_frag(smem, fb + 1, lane), ai);
        ao = WMMA_F16(aL[kc], lds_b_frag(smem, fb + 2, lane), ao);
        ao = WMMA_F16(aR[kc], lds_b_frag(smem, fb + 3, lane), ao);
        au = WMMA_F16(aL[kc], lds_b_frag(smem, fb + 4, lane), au);
        au = WMMA_F16(aR[kc], lds_b_frag(smem, fb + 5, lane), au);
        a1 = WMMA_F16(aL[kc], lds_b_frag(smem, fb + 6, lane), a1);
        a2 = WMMA_F16(aR[kc], lds_b_frag(smem, fb + 7, lane), a2);
      }
      float vbi = bUi[ncol], vbo = bUo[ncol], vbu = bUu[ncol];
      float vb1 = bf1[ncol], vb2 = bf2[ncol];
#pragma unroll
      for (int r = 0; r < 8; ++r) {
        int row = row0 + r + khalf * 8;
        if (row < nrows) {
          float iv = sigf(ai[r] + vbi);
          float ov = sigf(ao[r] + vbo);
          float uv = fmaxf(au[r] + vbu, 0.0f);
          float f1 = sigf(a1[r] + vb1);
          float f2 = sigf(a2[r] + vb2);
          float lc = c_prev[(size_t)(2 * row)     * HDIM + ncol];
          float rc = c_prev[(size_t)(2 * row + 1) * HDIM + ncol];
          float cv = iv * uv + f1 * lc + f2 * rc;
          float hv = ov * fmaxf(cv, 0.0f);
          c_out[(size_t)row * HDIM + ncol] = cv;
          h_out[(size_t)row * HDIM + ncol] = (_Float16)hv;
        }
      }
    }
  }
}

// ---- one large tree level (levels 1..TAIL_FIRST-1), multi-block ----
__global__ void __launch_bounds__(SBLOCK)
tree_level_kernel(_Float16* __restrict__ h_base, float* __restrict__ c_base,
                  int off_prev, int off_cur, int nrows,
                  const _Float16* __restrict__ UiT, const _Float16* __restrict__ UoT,
                  const _Float16* __restrict__ UuT,
                  const _Float16* __restrict__ Uf1T, const _Float16* __restrict__ Uf2T,
                  const float* __restrict__ bUi, const float* __restrict__ bUo,
                  const float* __restrict__ bUu,
                  const float* __restrict__ bf1, const float* __restrict__ bf2) {
  __shared__ _Float16 smem[32 * 512];   // 32KB stage
  int tile = blockIdx.x * SWAVES + (threadIdx.x >> 5);
  level_tile_body(smem, h_base, c_base, off_prev, off_cur, nrows, tile,
                  UiT, UoT, UuT, Uf1T, Uf2T, bUi, bUo, bUu, bf1, bf2);
}

// ---- fused tree tail (levels TAIL_FIRST..16): ONE workgroup, 8 dependent levels ----
// Each level has <= SWAVES tiles; agent fence + barrier between levels makes the
// previous level's global h/c stores visible across both CUs of the WGP.
__global__ void __launch_bounds__(SBLOCK)
tree_tail_kernel(_Float16* __restrict__ h_base, float* __restrict__ c_base,
                 int off_prev0, int nrows0, int nlevels,
                 const _Float16* __restrict__ UiT, const _Float16* __restrict__ UoT,
                 const _Float16* __restrict__ UuT,
                 const _Float16* __restrict__ Uf1T, const _Float16* __restrict__ Uf2T,
                 const float* __restrict__ bUi, const float* __restrict__ bUo,
                 const float* __restrict__ bUu,
                 const float* __restrict__ bf1, const float* __restrict__ bf2) {
  __shared__ _Float16 smem[32 * 512];
  int tile = threadIdx.x >> 5;     // single block: tile == wave id
  int off_prev = off_prev0;
  int n_prev   = nrows0;
  for (int lv = 0; lv < nlevels; ++lv) {
    int nrows   = n_prev >> 1;
    int off_cur = off_prev + n_prev;
    level_tile_body(smem, h_base, c_base, off_prev, off_cur, nrows, tile,
                    UiT, UoT, UuT, Uf1T, Uf2T, bUi, bUo, bUu, bf1, bf2);
    __threadfence();    // flush h/c stores to a CU-shared level before next level reads
    __syncthreads();
    off_prev = off_cur;
    n_prev   = nrows;
  }
}

// ---- projection: out = H @ P + bP, P padded 128x16 (NCLS=5) ----
__global__ void __launch_bounds__(PBLOCK)
tree_proj_kernel(const _Float16* __restrict__ h_base,
                 const _Float16* __restrict__ PT, const float* __restrict__ bP,
                 float* __restrict__ out) {
  int lane   = threadIdx.x & 31;
  int tile   = blockIdx.x * PWPB + (threadIdx.x >> 5);
  int ntiles = (TOTROWS + 15) >> 4;
  if (tile >= ntiles) return;
  int row0  = tile << 4;
  int m     = lane & 15;
  int khalf = lane >> 4;

  int arow = min(row0 + m, TOTROWS - 1);
  const _Float16* hrow = h_base + (size_t)arow * HDIM;
  v8f acc = {};
#pragma unroll
  for (int kc = 0; kc < 4; ++kc) {
    v16h a = *(const v16h*)(hrow + kc * 32 + khalf * 16);
    acc = WMMA_F16(a, load_b_frag(PT, m, kc * 32 + khalf * 16, HDIM), acc);
  }
  float vb = (m < NCLS) ? bP[m] : 0.0f;
#pragma unroll
  for (int r = 0; r < 8; ++r) {
    int row = row0 + r + khalf * 8;
    if (row < TOTROWS && m < NCLS)
      out[(size_t)row * NCLS + m] = acc[r] + vb;
  }
}

extern "C" void kernel_launch(void* const* d_in, const int* in_sizes, int n_in,
                              void* d_out, int out_size, void* d_ws, size_t ws_size,
                              hipStream_t stream) {
  (void)in_sizes; (void)n_in; (void)out_size; (void)ws_size;
  const int*   words = (const int*)  d_in[0];
  const float* emb   = (const float*)d_in[1];
  const float* Wi  = (const float*)d_in[2];  const float* bi  = (const float*)d_in[3];
  const float* Wo  = (const float*)d_in[4];  const float* bo  = (const float*)d_in[5];
  const float* Wu  = (const float*)d_in[6];  const float* bu  = (const float*)d_in[7];
  const float* Ui  = (const float*)d_in[8];  const float* bUi = (const float*)d_in[9];
  const float* Uo  = (const float*)d_in[10]; const float* bUo = (const float*)d_in[11];
  const float* Uu  = (const float*)d_in[12]; const float* bUu = (const float*)d_in[13];
  const float* Uf1 = (const float*)d_in[14]; const float* bf1 = (const float*)d_in[15];
  const float* Uf2 = (const float*)d_in[16]; const float* bf2 = (const float*)d_in[17];
  const float* P   = (const float*)d_in[18]; const float* bP  = (const float*)d_in[19];
  float* out = (float*)d_out;

  // workspace carve-up
  char* ws = (char*)d_ws;
  size_t off = 0;
  _Float16* h_base = (_Float16*)(ws + off); off += (size_t)131072 * HDIM * 2;
  float*    c_base = (float*)   (ws + off); off += (size_t)131072 * HDIM * 4;
  _Float16* WiT  = (_Float16*)(ws + off); off += 128 * 128 * 2;
  _Float16* WoT  = (_Float16*)(ws + off); off += 128 * 128 * 2;
  _Float16* WuT  = (_Float16*)(ws + off); off += 128 * 128 * 2;
  _Float16* UiT  = (_Float16*)(ws + off); off += 256 * 128 * 2;
  _Float16* UoT  = (_Float16*)(ws + off); off += 256 * 128 * 2;
  _Float16* UuT  = (_Float16*)(ws + off); off += 256 * 128 * 2;
  _Float16* Uf1T = (_Float16*)(ws + off); off += 128 * 128 * 2;
  _Float16* Uf2T = (_Float16*)(ws + off); off += 128 * 128 * 2;
  _Float16* PTp  = (_Float16*)(ws + off); off += 16 * 128 * 2;

  auto tr = [&](const float* src, _Float16* dst, int K, int N, int Npad) {
    int total = K * Npad;
    transpose_pad_kernel<<<(total + 255) / 256, 256, 0, stream>>>(src, dst, K, N, Npad);
  };
  tr(Wi,  WiT,  128, 128, 128);
  tr(Wo,  WoT,  128, 128, 128);
  tr(Wu,  WuT,  128, 128, 128);
  tr(Ui,  UiT,  256, 128, 128);
  tr(Uo,  UoT,  256, 128, 128);
  tr(Uu,  UuT,  256, 128, 128);
  tr(Uf1, Uf1T, 128, 128, 128);
  tr(Uf2, Uf2T, 128, 128, 128);
  tr(P,   PTp,  128, NCLS, 16);

  // leaves
  {
    int tiles = NLEAF / 16;
    tree_leaf_kernel<<<(tiles + SWAVES - 1) / SWAVES, SBLOCK, 0, stream>>>(
        words, emb, WiT, WoT, WuT, bi, bo, bu, h_base, c_base);
  }
  // large tree levels (serial dependency), one launch each
  int off_prev = 0, n_prev = NLEAF;
  for (int L = 1; L < TAIL_FIRST; ++L) {
    int n = NLEAF >> L;
    int off_cur = off_prev + n_prev;
    int tiles = (n + 15) / 16;
    int blocks = (tiles + SWAVES - 1) / SWAVES;
    tree_level_kernel<<<blocks, SBLOCK, 0, stream>>>(
        h_base, c_base, off_prev, off_cur, n,
        UiT, UoT, UuT, Uf1T, Uf2T, bUi, bUo, bUu, bf1, bf2);
    off_prev = off_cur; n_prev = n;
  }
  // fused tail: levels TAIL_FIRST..16 in one single-workgroup launch
  {
    int nlevels = 16 - (TAIL_FIRST - 1);
    tree_tail_kernel<<<1, SBLOCK, 0, stream>>>(
        h_base, c_base, off_prev, n_prev, nlevels,
        UiT, UoT, UuT, Uf1T, Uf2T, bUi, bUo, bUu, bf1, bf2);
  }
  // projection over all 131071 nodes
  {
    int tiles = (TOTROWS + 15) / 16;
    tree_proj_kernel<<<(tiles + PWPB - 1) / PWPB, PBLOCK, 0, stream>>>(h_base, PTp, bP, out);
  }
}
